// MultiHeadAttention_85959475462393
// MI455X (gfx1250) — matrix operand order repaired and checked by reading, not compiled, not run on hardware
//
#include <hip/hip_runtime.h>
#include <stdint.h>
#include <stddef.h>

// ---------------- problem constants (match reference) ----------------
constexpr int Bb = 2;
constexpr int Ss = 2048;
constexpr int Cc = 1024;
constexpr int Hh = 16;
constexpr int Dd = 64;          // head dim
constexpr int Mrows = Bb * Ss;  // 4096 flattened rows

// ---------------- WMMA types ----------------
typedef __bf16 v16bf __attribute__((ext_vector_type(16)));
typedef float  v8f   __attribute__((ext_vector_type(8)));
typedef int    v4i   __attribute__((ext_vector_type(4)));

union Frag { uint32_t u[8]; v16bf v; };

__device__ __forceinline__ uint16_t f2bf(float f) {
  uint32_t x = __builtin_bit_cast(uint32_t, f);
  x += 0x7fffu + ((x >> 16) & 1u);       // round-to-nearest-even
  return (uint16_t)(x >> 16);
}

__device__ __forceinline__ v8f wmma_bf16(const Frag& a, const Frag& b, v8f c) {
  return __builtin_amdgcn_wmma_f32_16x16x32_bf16(false, a.v, false, b.v,
                                                 (short)0, c, false, false);
}

// ---------------- async global->LDS staging (CDNA5), guarded ----------------
#if __has_builtin(__builtin_amdgcn_global_load_async_to_lds_b128) && \
    __has_builtin(__builtin_amdgcn_s_wait_asynccnt)
#define ATHENA_ASYNC 1
#endif

typedef __attribute__((address_space(1))) v4i* gptr_v4i;
typedef __attribute__((address_space(3))) v4i* lptr_v4i;

// copy 16 bytes from global to LDS (per-lane addresses)
__device__ __forceinline__ void copy16_to_lds(void* lds, const void* g) {
#ifdef ATHENA_ASYNC
  void* gnc = const_cast<void*>(g);
  __builtin_amdgcn_global_load_async_to_lds_b128(
      (gptr_v4i)gnc, (lptr_v4i)lds, /*offset=*/0, /*cpol=*/0);
#else
  *(uint4*)lds = *(const uint4*)g;
#endif
}

__device__ __forceinline__ void wait_async_copies() {
#ifdef ATHENA_ASYNC
  __builtin_amdgcn_s_wait_asynccnt(0);
#endif
}

// ---------------- f32 -> bf16 conversion ----------------
__global__ __launch_bounds__(256) void cvt_f32_bf16(const float* __restrict__ src,
                                                    uint16_t* __restrict__ dst, int n) {
  int i = (blockIdx.x * 256 + threadIdx.x) * 2;
  if (i < n) {
    uint32_t p = (uint32_t)f2bf(src[i]) | ((uint32_t)f2bf(src[i + 1]) << 16);
    *(uint32_t*)(dst + i) = p;
  }
}

// ---------------- tiled bf16 WMMA GEMM (double-buffered, pipelined) -------
// C[M,N] = A[M,K] * B[K,N];  A,B bf16 row-major.  Block tile 128x128, BK=32.
// 8 waves: waveM = wave&3 (32 rows), waveN = wave>>2 (64 cols); 2x4 tiles/wave.
template <bool OUT_BF16>
__global__ __launch_bounds__(256) void gemm_bf16(const uint16_t* __restrict__ A,
                                                 const uint16_t* __restrict__ Bm,
                                                 void* __restrict__ Cout,
                                                 int M, int N, int K, float scale) {
  __shared__ uint16_t As[2][128 * 32];   // [m][k]
  __shared__ uint16_t Bs[2][128 * 32];   // [n][k]  (transposed)

  const int tid  = threadIdx.x;
  const int lane = tid & 31;
  const int wave = tid >> 5;
  const int l15  = lane & 15;
  const int hf   = lane >> 4;

  const int bm = blockIdx.y * 128;
  const int bn = blockIdx.x * 128;
  const int wm = (wave & 3) * 32;     // 32 rows per wave
  const int wn = (wave >> 2) * 64;    // 64 cols per wave

  const v8f zero8 = {0.f, 0.f, 0.f, 0.f, 0.f, 0.f, 0.f, 0.f};
  v8f acc[2][4];
  #pragma unroll
  for (int mt = 0; mt < 2; ++mt)
    #pragma unroll
    for (int nt = 0; nt < 4; ++nt) acc[mt][nt] = zero8;

  // global-load assignments
  const int arow = tid >> 1;            // 0..127
  const int acol = (tid & 1) * 16;      // 0 or 16
  const int bk   = tid & 31;            // k in tile
  const int bcg  = (tid >> 5) * 16;     // col group of 16

  const int nsteps = K / 32;

  // ---- prologue: stage tile 0 into buffer 0 ----
  {
    const uint16_t* ag = A + (size_t)(bm + arow) * K + acol;
    copy16_to_lds(&As[0][arow * 32 + acol],     ag);
    copy16_to_lds(&As[0][arow * 32 + acol + 8], ag + 8);
    const uint16_t* bg = Bm + (size_t)bk * N + bn + bcg;
    uint4 b0 = *(const uint4*)bg;
    uint4 b1 = *(const uint4*)(bg + 8);
    const uint16_t* p0 = (const uint16_t*)&b0;
    const uint16_t* p1 = (const uint16_t*)&b1;
    #pragma unroll
    for (int j = 0; j < 8; ++j) Bs[0][(bcg + j) * 32 + bk]     = p0[j];
    #pragma unroll
    for (int j = 0; j < 8; ++j) Bs[0][(bcg + 8 + j) * 32 + bk] = p1[j];
  }

  for (int i = 0; i < nsteps; ++i) {
    const int cur = i & 1;
    const int nxt = cur ^ 1;
    wait_async_copies();
    __syncthreads();

    // ---- stage tile i+1: async A copies now, B global loads into regs ----
    const bool have_next = (i + 1) < nsteps;
    uint4 b0 = {}, b1 = {};
    if (have_next) {
      const int k0 = (i + 1) * 32;
      const uint16_t* ag = A + (size_t)(bm + arow) * K + k0 + acol;
      copy16_to_lds(&As[nxt][arow * 32 + acol],     ag);
      copy16_to_lds(&As[nxt][arow * 32 + acol + 8], ag + 8);
      const uint16_t* bg = Bm + (size_t)(k0 + bk) * N + bn + bcg;
      b0 = *(const uint4*)bg;
      b1 = *(const uint4*)(bg + 8);
    }

    // ---- compute on tile i (buffer cur): 8 WMMAs ----
    Frag af[2];
    #pragma unroll
    for (int mt = 0; mt < 2; ++mt) {
      const int row = wm + mt * 16 + l15;
      #pragma unroll
      for (int g = 0; g < 8; ++g) {
        const int kk = 2 * g + hf * 16;   // A layout
        af[mt].u[g] = *(const uint32_t*)&As[cur][row * 32 + kk];
      }
    }
    #pragma unroll
    for (int nt = 0; nt < 4; ++nt) {
      Frag bf;
      const int col = wn + nt * 16 + l15;
      #pragma unroll
      for (int g = 0; g < 8; ++g) {
        const int kk = 2 * g + hf * 16;                            // B layout
        bf.u[g] = *(const uint32_t*)&Bs[cur][col * 32 + kk];
      }
      #pragma unroll
      for (int mt = 0; mt < 2; ++mt)
        acc[mt][nt] = wmma_bf16(af[mt], bf, acc[mt][nt]);
    }

    // ---- finish staging: scatter B into Bs[nxt] (drained before barrier) ----
    if (have_next) {
      const uint16_t* p0 = (const uint16_t*)&b0;
      const uint16_t* p1 = (const uint16_t*)&b1;
      #pragma unroll
      for (int j = 0; j < 8; ++j) Bs[nxt][(bcg + j) * 32 + bk]     = p0[j];
      #pragma unroll
      for (int j = 0; j < 8; ++j) Bs[nxt][(bcg + 8 + j) * 32 + bk] = p1[j];
    }
  }

  // epilogue: C layout -> row = v + hf*8, col = l15
  #pragma unroll
  for (int mt = 0; mt < 2; ++mt)
    #pragma unroll
    for (int nt = 0; nt < 4; ++nt)
      #pragma unroll
      for (int v = 0; v < 8; ++v) {
        const int row = bm + wm + mt * 16 + hf * 8 + v;
        const int col = bn + wn + nt * 16 + l15;
        const float val = acc[mt][nt][v] * scale;
        if (OUT_BF16) ((uint16_t*)Cout)[(size_t)row * N + col] = f2bf(val);
        else          ((float*)Cout)[(size_t)row * N + col] = val;
      }
}

// ---------------- flash attention (causal, double-buffered K/V) -----------
// Grid: (S/128) * H * B blocks, 256 threads (8 waves).  Each wave: 16 q rows.
// q is pre-scaled by 1/sqrt(D).  kv holds K at col [0,C) and V at [C,2C).
__global__ __launch_bounds__(256) void attn_kernel(const uint16_t* __restrict__ q,
                                                   const uint16_t* __restrict__ kv,
                                                   uint16_t* __restrict__ out) {
  constexpr int NQB = Ss / 128;
  const int bx = blockIdx.x;
  const int qb = bx % NQB;
  const int h  = (bx / NQB) % Hh;
  const int b  = bx / (NQB * Hh);
  const int q0 = qb * 128;

  const int tid  = threadIdx.x;
  const int lane = tid & 31;
  const int w    = tid >> 5;
  const int l15  = lane & 15;
  const int hf   = lane >> 4;

  __shared__ uint16_t Ks[2][64 * 64];   // [key][d]
  __shared__ uint16_t Vs[2][64 * 64];   // [d][key] (transposed)
  __shared__ uint16_t Ps[8][16 * 64];   // per-wave P staging (wave-private)

  // Q fragments (2 d-steps of 32), loaded once
  Frag qa[2];
  {
    const int qrow = q0 + w * 16 + l15;
    const uint16_t* qb_ptr = q + (size_t)(b * Ss + qrow) * Cc + h * Dd;
    #pragma unroll
    for (int ks = 0; ks < 2; ++ks)
      #pragma unroll
      for (int g = 0; g < 8; ++g) {
        const int kk = 2 * g + hf * 16 + ks * 32;
        qa[ks].u[g] = *(const uint32_t*)(qb_ptr + kk);
      }
  }

  const v8f zero8 = {0.f, 0.f, 0.f, 0.f, 0.f, 0.f, 0.f, 0.f};
  v8f o[4]; o[0] = zero8; o[1] = zero8; o[2] = zero8; o[3] = zero8;
  float m[8], lsum[8];
  #pragma unroll
  for (int v = 0; v < 8; ++v) { m[v] = -3.0e38f; lsum[v] = 0.f; }

  // K/V tile load assignment: 16 bf16 per thread
  const int r  = tid >> 2;         // key row 0..63
  const int db = (tid & 3) * 16;   // d base

  const int nkb = q0 / 64 + 2;     // key blocks needed for causal coverage

  // ---- prologue: stage key block 0 into buffer 0 ----
  {
    const uint16_t* kg = kv + (size_t)(b * Ss + r) * (2 * Cc) + h * Dd + db;
    copy16_to_lds(&Ks[0][r * 64 + db],     kg);
    copy16_to_lds(&Ks[0][r * 64 + db + 8], kg + 8);
    const uint16_t* vg = kv + (size_t)(b * Ss + r) * (2 * Cc) + Cc + h * Dd + db;
    uint4 v0 = *(const uint4*)vg;
    uint4 v1 = *(const uint4*)(vg + 8);
    const uint16_t* p0 = (const uint16_t*)&v0;
    const uint16_t* p1 = (const uint16_t*)&v1;
    #pragma unroll
    for (int j = 0; j < 8; ++j) Vs[0][(db + j) * 64 + r]     = p0[j];
    #pragma unroll
    for (int j = 0; j < 8; ++j) Vs[0][(db + 8 + j) * 64 + r] = p1[j];
  }

  for (int i = 0; i < nkb; ++i) {
    const int kb  = i * 64;
    const int cur = i & 1;
    const int nxt = cur ^ 1;
    wait_async_copies();
    __syncthreads();

    // ---- stage key block i+1: async K now, V global loads into regs ----
    const bool have_next = (i + 1) < nkb;
    uint4 v0 = {}, v1 = {};
    if (have_next) {
      const int kbn = kb + 64;
      const uint16_t* kg = kv + (size_t)(b * Ss + kbn + r) * (2 * Cc) + h * Dd + db;
      copy16_to_lds(&Ks[nxt][r * 64 + db],     kg);
      copy16_to_lds(&Ks[nxt][r * 64 + db + 8], kg + 8);
      const uint16_t* vg = kv + (size_t)(b * Ss + kbn + r) * (2 * Cc) + Cc + h * Dd + db;
      v0 = *(const uint4*)vg;
      v1 = *(const uint4*)(vg + 8);
    }

    // ---- scores: S = Q * K^T (16x64 per wave) ----
    v8f s[4];
    #pragma unroll
    for (int nt = 0; nt < 4; ++nt) {
      Frag bkf;
      #pragma unroll
      for (int g = 0; g < 8; ++g)
        bkf.u[g] = *(const uint32_t*)&Ks[cur][(nt * 16 + l15) * 64 + 2 * g + hf * 16];
      s[nt] = wmma_bf16(qa[0], bkf, zero8);
      #pragma unroll
      for (int g = 0; g < 8; ++g)
        bkf.u[g] = *(const uint32_t*)&Ks[cur][(nt * 16 + l15) * 64 + 2 * g + hf * 16 + 32];
      s[nt] = wmma_bf16(qa[1], bkf, s[nt]);
    }

    // ---- causal mask (skip when whole block is below the diagonal) ----
    if (kb + 63 > q0 + w * 16) {
      #pragma unroll
      for (int nt = 0; nt < 4; ++nt) {
        const int key = kb + nt * 16 + l15;
        #pragma unroll
        for (int v = 0; v < 8; ++v) {
          const int qr = q0 + w * 16 + hf * 8 + v;
          if (key > qr) s[nt][v] = -1.0e9f;
        }
      }
    }

    // ---- online softmax ----
    float rm[8];
    #pragma unroll
    for (int v = 0; v < 8; ++v)
      rm[v] = fmaxf(fmaxf(s[0][v], s[1][v]), fmaxf(s[2][v], s[3][v]));
    #pragma unroll
    for (int msk = 8; msk >= 1; msk >>= 1)
      #pragma unroll
      for (int v = 0; v < 8; ++v)
        rm[v] = fmaxf(rm[v], __shfl_xor(rm[v], msk, 32));

    float al[8];
    #pragma unroll
    for (int v = 0; v < 8; ++v) {
      const float mn = fmaxf(m[v], rm[v]);
      al[v] = __expf(m[v] - mn);
      m[v]  = mn;
    }
    float rs[8];
    #pragma unroll
    for (int v = 0; v < 8; ++v) rs[v] = 0.f;
    #pragma unroll
    for (int nt = 0; nt < 4; ++nt)
      #pragma unroll
      for (int v = 0; v < 8; ++v) {
        const float p = __expf(s[nt][v] - m[v]);
        s[nt][v] = p;
        rs[v] += p;
      }
    #pragma unroll
    for (int msk = 8; msk >= 1; msk >>= 1)
      #pragma unroll
      for (int v = 0; v < 8; ++v)
        rs[v] += __shfl_xor(rs[v], msk, 32);
    #pragma unroll
    for (int v = 0; v < 8; ++v) lsum[v] = lsum[v] * al[v] + rs[v];
    #pragma unroll
    for (int nt = 0; nt < 4; ++nt)
      #pragma unroll
      for (int v = 0; v < 8; ++v) o[nt][v] *= al[v];

    // ---- finish staging: scatter V into Vs[nxt] ----
    if (have_next) {
      const uint16_t* p0 = (const uint16_t*)&v0;
      const uint16_t* p1 = (const uint16_t*)&v1;
      #pragma unroll
      for (int j = 0; j < 8; ++j) Vs[nxt][(db + j) * 64 + r]     = p0[j];
      #pragma unroll
      for (int j = 0; j < 8; ++j) Vs[nxt][(db + 8 + j) * 64 + r] = p1[j];
    }

    // ---- stage P (bf16): wave-private LDS, DS ops are in-order per wave ----
    #pragma unroll
    for (int nt = 0; nt < 4; ++nt)
      #pragma unroll
      for (int v = 0; v < 8; ++v)
        Ps[w][(hf * 8 + v) * 64 + nt * 16 + l15] = f2bf(s[nt][v]);

    // ---- O += P * V ----
    #pragma unroll
    for (int ks = 0; ks < 2; ++ks) {
      Frag pa;
      #pragma unroll
      for (int g = 0; g < 8; ++g) {
        const int kk = 2 * g + hf * 16 + ks * 32;
        pa.u[g] = *(const uint32_t*)&Ps[w][l15 * 64 + kk];
      }
      #pragma unroll
      for (int nt = 0; nt < 4; ++nt) {
        Frag vb;
        #pragma unroll
        for (int g = 0; g < 8; ++g)
          vb.u[g] = *(const uint32_t*)&Vs[cur][(nt * 16 + l15) * 64 + 2 * g + hf * 16 + ks * 32];
        o[nt] = wmma_bf16(pa, vb, o[nt]);
      }
    }
  }

  // ---- normalize and store (bf16, [b, q, h*D + d]) ----
  #pragma unroll
  for (int nt = 0; nt < 4; ++nt)
    #pragma unroll
    for (int v = 0; v < 8; ++v) {
      const int qr = q0 + w * 16 + hf * 8 + v;
      const float val = o[nt][v] / lsum[v];
      out[(size_t)(b * Ss + qr) * Cc + h * Dd + nt * 16 + l15] = f2bf(val);
    }
}

// ---------------- host launcher ----------------
extern "C" void kernel_launch(void* const* d_in, const int* in_sizes, int n_in,
                              void* d_out, int out_size, void* d_ws, size_t ws_size,
                              hipStream_t stream) {
  const float* query = (const float*)d_in[0];
  const float* memry = (const float*)d_in[1];
  const float* Wq    = (const float*)d_in[2];
  const float* Wkv   = (const float*)d_in[3];
  const float* Wo    = (const float*)d_in[4];
  // d_in[5] = mask (bool) — causal, handled analytically.

  uint8_t* ws = (uint8_t*)d_ws;
  const size_t N_QRY  = (size_t)Mrows * Cc;       // 4M
  const size_t N_WQ   = (size_t)Cc * Cc;          // 1M
  const size_t N_WKV  = (size_t)Cc * 2 * Cc;      // 2M

  uint16_t* query_bf = (uint16_t*)(ws);                          // 8 MB
  uint16_t* memry_bf = (uint16_t*)(ws + 8u  * 1024 * 1024);      // 8 MB
  uint16_t* Wq_bf    = (uint16_t*)(ws + 16u * 1024 * 1024);      // 2 MB
  uint16_t* Wkv_bf   = (uint16_t*)(ws + 18u * 1024 * 1024);      // 4 MB
  uint16_t* Wo_bf    = (uint16_t*)(ws + 22u * 1024 * 1024);      // 2 MB
  uint16_t* q_bf     = (uint16_t*)(ws + 24u * 1024 * 1024);      // 8 MB
  uint16_t* kv_bf    = (uint16_t*)(ws + 32u * 1024 * 1024);      // 16 MB
  uint16_t* attn_bf  = (uint16_t*)(ws + 48u * 1024 * 1024);      // 8 MB

  auto cvt = [&](const float* s, uint16_t* d, size_t n) {
    cvt_f32_bf16<<<dim3((unsigned)((n / 2 + 255) / 256)), 256, 0, stream>>>(s, d, (int)n);
  };
  cvt(query, query_bf, N_QRY);
  cvt(memry, memry_bf, N_QRY);
  cvt(Wq,    Wq_bf,    N_WQ);
  cvt(Wkv,   Wkv_bf,   N_WKV);
  cvt(Wo,    Wo_bf,    N_WQ);

  const float inv_sqrt_d = 0.125f;   // 1/sqrt(64), folded into q

  // q = query @ Wq  (scaled)
  gemm_bf16<true><<<dim3(Cc / 128, Mrows / 128), 256, 0, stream>>>(
      query_bf, Wq_bf, q_bf, Mrows, Cc, Cc, inv_sqrt_d);
  // kv = memory @ Wkv
  gemm_bf16<true><<<dim3(2 * Cc / 128, Mrows / 128), 256, 0, stream>>>(
      memry_bf, Wkv_bf, kv_bf, Mrows, 2 * Cc, Cc, 1.0f);
  // flash attention
  attn_kernel<<<dim3((Ss / 128) * Hh * Bb), 256, 0, stream>>>(q_bf, kv_bf, attn_bf);
  // out = attn @ Wo  (f32 to d_out)
  gemm_bf16<false><<<dim3(Cc / 128, Mrows / 128), 256, 0, stream>>>(
      attn_bf, Wo_bf, d_out, Mrows, Cc, Cc, 1.0f);

  (void)in_sizes; (void)n_in; (void)out_size; (void)ws_size;
}
